// W8A8OF16LinearStaticScale_9878424781509
// MI455X (gfx1250) — 8-bit matrix operand order repaired and checked by reading, compiled, not run on hardware
//
#include <hip/hip_runtime.h>

typedef __attribute__((ext_vector_type(8))) int v8i;
typedef __attribute__((ext_vector_type(4))) int v4i;
typedef __attribute__((ext_vector_type(2))) int v2i;

#define TOKENS 8192
#define IN_F   4096
#define OUT_F  4096

#define BM 128
#define BN 128
#define KSTEP 64

#define ROW_STRIDE 80
#define TILE_BYTES (128 * ROW_STRIDE)
#define STAGE_BYTES (2 * TILE_BYTES)
#define NKT (IN_F / KSTEP)

__global__ __launch_bounds__(256) void w8a8_wmma_kernel(
    const signed char* __restrict__ x,
    const signed char* __restrict__ w,
    const _Float16*  __restrict__ scale,
    const _Float16*  __restrict__ bias,
    _Float16*        __restrict__ out)
{
    __shared__ __align__(16) signed char smem[2 * STAGE_BYTES];

    const int tid    = threadIdx.x;
    const int lane   = tid & 31;
    const int wave   = tid >> 5;
    const int wave_m = wave & 1;
    const int wave_n = wave >> 1;

    const int blkM = blockIdx.y * BM;
    const int blkN = blockIdx.x * BN;

    const int l16 = lane & 15;
    const int lhi = lane >> 4;

    const int ci0   = tid;
    const int row0  = ci0 >> 2;
    const int col0  = (ci0 & 3) * 16;
    const int ci1   = tid + 256;
    const int row1  = ci1 >> 2;
    const int col1  = (ci1 & 3) * 16;

    const signed char* gA0 = x + (size_t)(blkM + row0) * IN_F + col0;
    const signed char* gA1 = x + (size_t)(blkM + row1) * IN_F + col1;
    const signed char* gB0 = w + (size_t)(blkN + row0) * IN_F + col0;
    const signed char* gB1 = w + (size_t)(blkN + row1) * IN_F + col1;

    const unsigned sbase = (unsigned)(size_t)&smem[0];
    const unsigned lA0 = sbase + row0 * ROW_STRIDE + col0;
    const unsigned lA1 = sbase + row1 * ROW_STRIDE + col1;
    const unsigned lB0 = sbase + TILE_BYTES + row0 * ROW_STRIDE + col0;
    const unsigned lB1 = sbase + TILE_BYTES + row1 * ROW_STRIDE + col1;

    auto stage = [&](int k, unsigned bufoff) {
        asm volatile("global_load_async_to_lds_b128 %0, %1, off"
                     :: "v"(lA0 + bufoff), "v"(gA0 + k) : "memory");
        asm volatile("global_load_async_to_lds_b128 %0, %1, off"
                     :: "v"(lB0 + bufoff), "v"(gB0 + k) : "memory");
        asm volatile("global_load_async_to_lds_b128 %0, %1, off"
                     :: "v"(lA1 + bufoff), "v"(gA1 + k) : "memory");
        asm volatile("global_load_async_to_lds_b128 %0, %1, off"
                     :: "v"(lB1 + bufoff), "v"(gB1 + k) : "memory");
    };

    v8i acc[4][2];
#pragma unroll
    for (int mt = 0; mt < 4; ++mt)
#pragma unroll
        for (int nt = 0; nt < 2; ++nt)
#pragma unroll
            for (int e = 0; e < 8; ++e)
                acc[mt][nt][e] = 0;

    stage(0, 0);

    for (int kt = 0; kt < NKT; ++kt) {
        const int cur = kt & 1;
        const unsigned curoff = cur ? STAGE_BYTES : 0;

        if (kt + 1 < NKT) {
            stage((kt + 1) * KSTEP, cur ? 0 : STAGE_BYTES);
            asm volatile("s_wait_asynccnt 0x4" ::: "memory");
        } else {
            asm volatile("s_wait_asynccnt 0x0" ::: "memory");
        }
        __syncthreads();

        const signed char* sA = &smem[curoff];
        const signed char* sB = &smem[curoff + TILE_BYTES];

        v8i a[4];
#pragma unroll
        for (int mt = 0; mt < 4; ++mt) {
            const signed char* base =
                sA + (wave_m * 64 + mt * 16 + l16) * ROW_STRIDE + lhi * 16;
            v4i p0 = *(const v4i*)(base +  0);
            v4i p1 = *(const v4i*)(base + 32);
            a[mt] = (v8i){p0.x, p0.y, p0.z, p0.w, p1.x, p1.y, p1.z, p1.w};
        }

        v8i b[2];
#pragma unroll
        for (int nt = 0; nt < 2; ++nt) {
            const signed char* base =
                sB + (wave_n * 32 + nt * 16 + l16) * ROW_STRIDE + lhi * 16;
            v4i q0 = *(const v4i*)(base +  0);
            v4i q1 = *(const v4i*)(base + 32);
            b[nt] = (v8i){q0.x, q0.y, q0.z, q0.w, q1.x, q1.y, q1.z, q1.w};
        }

#pragma unroll
        for (int mt = 0; mt < 4; ++mt)
#pragma unroll
            for (int nt = 0; nt < 2; ++nt)
                acc[mt][nt] = __builtin_amdgcn_wmma_i32_16x16x64_iu8(
                     true, a[mt],
                     true, b[nt],
                    acc[mt][nt],
                     false,  false);

        __syncthreads();
    }

#pragma unroll
    for (int nt = 0; nt < 2; ++nt) {
        const int col = blkN + wave_n * 32 + nt * 16 + l16;
        const float s = (float)scale[col];
        const _Float16 bz = bias[col];
#pragma unroll
        for (int mt = 0; mt < 4; ++mt) {
            const int rowbase = blkM + wave_m * 64 + mt * 16 + lhi * 8;
#pragma unroll
            for (int r = 0; r < 8; ++r) {
                float f = (float)acc[mt][nt][r] * s;
                _Float16 h = (_Float16)f;
                h = h + bz;
                __builtin_nontemporal_store(h, &out[(size_t)(rowbase + r) * OUT_F + col]);
            }
        }
    }
}

__global__ void bias_tail_kernel(const _Float16* __restrict__ bias,
                                 _Float16* __restrict__ out_tail)
{
    int i = blockIdx.x * blockDim.x + threadIdx.x;
    if (i < OUT_F) out_tail[i] = bias[i];
}

extern "C" void kernel_launch(void* const* d_in, const int* in_sizes, int n_in,
                              void* d_out, int out_size, void* d_ws, size_t ws_size,
                              hipStream_t stream) {
    const signed char* x     = (const signed char*)d_in[0];
    const signed char* w     = (const signed char*)d_in[1];
    const _Float16*    scale = (const _Float16*)d_in[2];
    const _Float16*    bias  = (const _Float16*)d_in[3];
    _Float16*          out   = (_Float16*)d_out;

    dim3 grid(OUT_F / BN, TOKENS / BM);
    w8a8_wmma_kernel<<<grid, 256, 0, stream>>>(x, w, scale, bias, out);

    bias_tail_kernel<<<OUT_F / 256, 256, 0, stream>>>(
        bias, out + (size_t)TOKENS * OUT_F);
}
